// RewYolov2_61280593380131
// MI455X (gfx1250) — compile-verified, matrix operand order checked by reading, not run on hardware
//
#include <hip/hip_runtime.h>
#include <hip/hip_bf16.h>

typedef __attribute__((ext_vector_type(16))) _Float16 v16h;
typedef __attribute__((ext_vector_type(8)))  _Float16 v8h;
typedef __attribute__((ext_vector_type(8)))  float    v8f;
typedef int v4i __attribute__((vector_size(16)));

#define EPSV   1e-5f
#define SLOPE  0.1f

#if defined(__has_builtin)
#if __has_builtin(__builtin_amdgcn_global_load_async_to_lds_b128) && \
    __has_builtin(__builtin_amdgcn_s_wait_asynccnt)
#define USE_ASYNC 1
#endif
#endif

#define GAS __attribute__((address_space(1)))
#define LAS __attribute__((address_space(3)))

// ---------------- problem sizes ----------------
constexpr int BB    = 16;
constexpr int H1    = 38, W1 = 38, C1IN = 512, C1OUT = 64;
constexpr int NPIX1 = BB * H1 * W1;            // 23104 = 1444 * 16
constexpr int H2    = 19, W2D = 19, C2IN = 1280, C2OUT = 1024;
constexpr int NPIX2 = BB * H2 * W2D;           // 5776 = 361 * 16

constexpr int LDS_STRIDE = C2IN + 8;           // 1288 halfs: pad -> conflict-free banks

// ---------------- workspace layout (bytes) ----------------
constexpr size_t OFF_FT   = 0;                                         // feat1 NHWC f16
constexpr size_t SZ_FT    = (size_t)NPIX1 * C1IN * 2;                  // 23,658,496
constexpr size_t OFF_CAT  = OFF_FT + SZ_FT;
constexpr size_t SZ_CAT   = (size_t)(NPIX2 + 1) * C2IN * 2;            // +1 ghost zero row
constexpr size_t OFF_W1P  = OFF_CAT + SZ_CAT;
constexpr size_t SZ_W1P   = (size_t)4 * 16 * 32 * 16 * 2;              // 65,536
constexpr size_t OFF_W2P  = OFF_W1P + SZ_W1P;
constexpr size_t SZ_W2P   = (size_t)64 * 9 * 40 * 32 * 16 * 2;         // 23,592,960
constexpr size_t OFF_PRE  = OFF_W2P + SZ_W2P;                          // pre f32: 23,658,496

__device__ __forceinline__ v16h join8(v8h lo, v8h hi) {
  return __builtin_shufflevector(lo, hi, 0,1,2,3,4,5,6,7,8,9,10,11,12,13,14,15);
}

// -------- feat1 NCHW f32 -> NHWC f16 --------
__global__ void k_t1(const float* __restrict__ feat1, _Float16* __restrict__ ft) {
  int idx = blockIdx.x * 256 + threadIdx.x;
  if (idx >= BB * C1IN * H1 * W1) return;
  int pix = idx % (H1 * W1);
  int t   = idx / (H1 * W1);
  int c   = t % C1IN;
  int b   = t / C1IN;
  ft[(b * (H1 * W1) + pix) * C1IN + c] = (_Float16)feat1[idx];
}

// -------- feat0 NCHW f32 -> cat[.., 256:1280] NHWC f16 --------
__global__ void k_t0(const float* __restrict__ feat0, _Float16* __restrict__ cat) {
  int idx = blockIdx.x * 256 + threadIdx.x;
  if (idx >= BB * 1024 * (H2 * W2D)) return;
  int pix = idx % (H2 * W2D);
  int t   = idx / (H2 * W2D);
  int c   = t % 1024;
  int b   = t / 1024;
  cat[(b * (H2 * W2D) + pix) * C2IN + 256 + c] = (_Float16)feat0[idx];
}

// -------- zero the ghost (pad) row of cat --------
__global__ void k_zero(_Float16* __restrict__ cat) {
  int t = blockIdx.x * 256 + threadIdx.x;
  if (t < C2IN) cat[(size_t)NPIX2 * C2IN + t] = (_Float16)0.0f;
}

// -------- pack w1 (64,512,1,1) f32 -> B-fragment tiles f16 --------
__global__ void k_packw1(const float* __restrict__ w1, _Float16* __restrict__ w1p) {
  int t = blockIdx.x * 256 + threadIdx.x;
  if (t >= 4 * 16 * 32) return;
  int lane  = t & 31;
  int chunk = (t >> 5) & 15;
  int nt    = t >> 9;
  int oc    = nt * 16 + (lane & 15);
  int kb    = (lane >> 4) * 8;
  _Float16* dst = w1p + t * 16;
  #pragma unroll
  for (int e = 0; e < 16; ++e) {
    int k = chunk * 32 + kb + (e & 7) + ((e >> 3) << 4);
    dst[e] = (_Float16)w1[oc * C1IN + k];
  }
}

// -------- pack w2 (1024,1280,3,3) f32 -> B-fragment tiles f16 --------
// layout: [ntile(64)][kh(3)][kw(3)][chunk(40)][lane(32)][e(16)]
__global__ void k_packw2(const float* __restrict__ w2, _Float16* __restrict__ w2p) {
  int t = blockIdx.x * 256 + threadIdx.x;
  if (t >= 64 * 9 * 40 * 32) return;
  int lane  = t & 31;
  int chunk = (t >> 5) % 40;
  int rest  = t / (32 * 40);
  int kw    = rest % 3;
  int kh    = (rest / 3) % 3;
  int nt    = rest / 9;
  int oc    = nt * 16 + (lane & 15);
  int kb    = (lane >> 4) * 8;
  _Float16* dst = w2p + t * 16;
  #pragma unroll
  for (int e = 0; e < 16; ++e) {
    int c = chunk * 32 + kb + (e & 7) + ((e >> 3) << 4);
    dst[e] = (_Float16)w2[((oc * C2IN + c) * 3 + kh) * 3 + kw];
  }
}

// -------- conv1 1x1 (512->64) GEMM + BN + leaky + reorg scatter into cat --------
__global__ void __launch_bounds__(128) k_conv1(
    const _Float16* __restrict__ ft, const _Float16* __restrict__ w1p,
    const float* __restrict__ g1, const float* __restrict__ b1,
    const float* __restrict__ m1, const float* __restrict__ v1,
    _Float16* __restrict__ cat)
{
  int lane  = threadIdx.x & 31;
  int nt    = threadIdx.x >> 5;         // 0..3  (N tile = wave id)
  int mtile = blockIdx.x;               // 0..1443
  int hi8   = lane >> 4;
  int p     = mtile * 16 + (lane & 15); // A row pixel

  const _Float16* abase = ft + (size_t)p * C1IN + hi8 * 8;
  const _Float16* bbase = w1p + ((size_t)(nt * 16) * 32 + lane) * 16;

  v8f acc = {};
  #pragma unroll 4
  for (int kk = 0; kk < 16; ++kk) {
    v8h  alo = *(const v8h*)(abase + kk * 32);
    v8h  ahi = *(const v8h*)(abase + kk * 32 + 16);
    v16h a   = join8(alo, ahi);
    v16h b   = *(const v16h*)(bbase + kk * 512);
    acc = __builtin_amdgcn_wmma_f32_16x16x32_f16(false, a, false, b,
                                                 (short)0, acc, false, false);
  }

  int   oc = nt * 16 + (lane & 15);
  float sc = g1[oc] * rsqrtf(v1[oc] + EPSV);
  float sh = b1[oc] - m1[oc] * sc;

  #pragma unroll
  for (int r = 0; r < 8; ++r) {
    int   pp  = mtile * 16 + r + hi8 * 8;
    float x   = acc[r] * sc + sh;
    x = (x > 0.0f) ? x : SLOPE * x;
    int b_  = pp / (H1 * W1);
    int rem = pp % (H1 * W1);
    int y   = rem / W1, xx = rem % W1;
    int hh  = y >> 1,  ww = xx >> 1;
    int sb  = (y & 1) * 2 + (xx & 1);   // reorg sub-block id
    cat[((size_t)(b_ * (H2 * W2D) + hh * W2D + ww)) * C2IN + sb * 64 + oc] = (_Float16)x;
  }
}

__device__ __forceinline__ void bn_store(v8f acc, int oc, int mtile, int hi8,
    const float* __restrict__ g2, const float* __restrict__ b2,
    const float* __restrict__ m2, const float* __restrict__ v2,
    float* __restrict__ pre)
{
  float sc = g2[oc] * rsqrtf(v2[oc] + EPSV);
  float sh = b2[oc] - m2[oc] * sc;
  #pragma unroll
  for (int r = 0; r < 8; ++r) {
    int   pp = mtile * 16 + r + hi8 * 8;
    float xv = acc[r] * sc + sh;
    xv = (xv > 0.0f) ? xv : SLOPE * xv;
    pre[(size_t)pp * C2OUT + oc] = xv;
  }
}

// -------- conv2 3x3 (1280->1024) implicit GEMM, LDS-staged A, 4 N-tiles/wave --------
// grid(361, 2), block 256 = 8 waves; block tile = 16 pixels x 512 channels.
__global__ void __launch_bounds__(256) k_conv2(
    const _Float16* __restrict__ cat, const _Float16* __restrict__ w2p,
    const float* __restrict__ g2, const float* __restrict__ b2,
    const float* __restrict__ m2, const float* __restrict__ v2,
    float* __restrict__ pre)
{
  __shared__ _Float16 shA[16 * LDS_STRIDE];    // 41,216 B

  int lane  = threadIdx.x & 31;
  int wv    = threadIdx.x >> 5;                // 0..7
  int mtile = blockIdx.x;                      // 0..360
  int hi8   = lane >> 4;

  // ---- staging role: 16 threads per pixel row ----
  int srow = threadIdx.x >> 4;                 // 0..15
  int sseg = threadIdx.x & 15;                 // 0..15  (80 halfs each)
  int sp   = mtile * 16 + srow;                // pixel for this staging row (<5776)
  int sb_  = sp / (H2 * W2D);
  int srem = sp % (H2 * W2D);
  int sy   = srem / W2D;
  int sx   = srem % W2D;
  const _Float16* ghost = cat + (size_t)NPIX2 * C2IN;

  // ---- compute role: 4 consecutive N tiles per wave ----
  int ntile0 = blockIdx.y * 32 + wv * 4;       // 0..60
  const _Float16* bb0 = w2p + ((size_t)(ntile0 + 0) * 9 * 40 * 32 + lane) * 16;
  const _Float16* bb1 = w2p + ((size_t)(ntile0 + 1) * 9 * 40 * 32 + lane) * 16;
  const _Float16* bb2 = w2p + ((size_t)(ntile0 + 2) * 9 * 40 * 32 + lane) * 16;
  const _Float16* bb3 = w2p + ((size_t)(ntile0 + 3) * 9 * 40 * 32 + lane) * 16;
  const _Float16* arow = shA + (lane & 15) * LDS_STRIDE + hi8 * 8;

  v8f ac0 = {}, ac1 = {}, ac2 = {}, ac3 = {};

  for (int kh = 0; kh < 3; ++kh) {
    int yy = sy + kh - 1;
    for (int kw = 0; kw < 3; ++kw) {
      int  xx = sx + kw - 1;
      bool ok = (yy >= 0) & (yy < H2) & (xx >= 0) & (xx < W2D);
      const _Float16* gbase = ok
          ? cat + ((size_t)(sb_ * (H2 * W2D) + yy * W2D + xx)) * C2IN
          : ghost;
      _Float16* lbase = shA + srow * LDS_STRIDE;
      // stage 16 x 1280 halfs (40 KB) into LDS: 10 x 16B per thread
      #pragma unroll
      for (int i = 0; i < 10; ++i) {
        int off = sseg * 80 + i * 8;
#ifdef USE_ASYNC
        __builtin_amdgcn_global_load_async_to_lds_b128(
            (GAS v4i*)(gbase + off), (LAS v4i*)(lbase + off), 0, 0);
#else
        v8h d = *(const v8h*)(gbase + off);
        *(v8h*)(lbase + off) = d;
#endif
      }
#ifdef USE_ASYNC
      __builtin_amdgcn_s_wait_asynccnt(0);
#endif
      __syncthreads();

      int kp = (kh * 3 + kw) * (40 * 512);
      const _Float16* bc0 = bb0 + kp;
      const _Float16* bc1 = bb1 + kp;
      const _Float16* bc2 = bb2 + kp;
      const _Float16* bc3 = bb3 + kp;
      __builtin_prefetch(bc0 + 40 * 512, 0, 1);   // next (kh,kw) weight block

      #pragma unroll 2
      for (int kk = 0; kk < 40; ++kk) {
        v8h  alo = *(const v8h*)(arow + kk * 32);
        v8h  ahi = *(const v8h*)(arow + kk * 32 + 16);
        v16h a   = join8(alo, ahi);
        v16h bf0 = *(const v16h*)(bc0 + kk * 512);
        v16h bf1 = *(const v16h*)(bc1 + kk * 512);
        v16h bf2 = *(const v16h*)(bc2 + kk * 512);
        v16h bf3 = *(const v16h*)(bc3 + kk * 512);
        ac0 = __builtin_amdgcn_wmma_f32_16x16x32_f16(false, a, false, bf0, (short)0, ac0, false, false);
        ac1 = __builtin_amdgcn_wmma_f32_16x16x32_f16(false, a, false, bf1, (short)0, ac1, false, false);
        ac2 = __builtin_amdgcn_wmma_f32_16x16x32_f16(false, a, false, bf2, (short)0, ac2, false, false);
        ac3 = __builtin_amdgcn_wmma_f32_16x16x32_f16(false, a, false, bf3, (short)0, ac3, false, false);
      }
      __syncthreads();
    }
  }

  int ocb = ntile0 * 16 + (lane & 15);
  bn_store(ac0, ocb +  0, mtile, hi8, g2, b2, m2, v2, pre);
  bn_store(ac1, ocb + 16, mtile, hi8, g2, b2, m2, v2, pre);
  bn_store(ac2, ocb + 32, mtile, hi8, g2, b2, m2, v2, pre);
  bn_store(ac3, ocb + 48, mtile, hi8, g2, b2, m2, v2, pre);
}

// -------- decode: masked 1x1 conv + argmax + gather --------
__global__ void __launch_bounds__(128) k_decode(
    const float* __restrict__ pre, const float* __restrict__ wd,
    const float* __restrict__ bd, float* __restrict__ out)
{
  int t = blockIdx.x * 128 + threadIdx.x;
  if (t >= NPIX2 * 5) return;
  int a = t / NPIX2;
  int p = t % NPIX2;
  const float* pr = pre + (size_t)p * C2OUT;
  const float* w5 = wd + (a * 6 + 5) * C2OUT;
  float bias5 = bd[a * 6 + 5];

  float cls[20];
  float best = -3.4e38f;
  int   ids  = 0;
  #pragma unroll
  for (int n = 0; n < 20; ++n) {
    float s = 0.0f;
    for (int c = n; c < C2OUT; c += 20) s += w5[c] * pr[c];
    s += bias5;
    cls[n] = s;
    if (s > best) { best = s; ids = n; }   // first-max, matches jnp.argmax
  }

  float box[5];
  #pragma unroll
  for (int j = 0; j < 5; ++j) {
    const float* wj = wd + (a * 6 + j) * C2OUT;
    float s = bd[a * 6 + j];
    for (int c = ids; c < C2OUT; c += 20) s += wj[c] * pr[c];
    box[j] = s;
  }

  int b_ = p / (H2 * W2D);
  int hw = p % (H2 * W2D);
  float* ob = out + (size_t)b_ * 125 * (H2 * W2D) + hw;
  #pragma unroll
  for (int j = 0; j < 5; ++j)  ob[(a * 25 + j) * (H2 * W2D)]      = box[j];
  #pragma unroll
  for (int n = 0; n < 20; ++n) ob[(a * 25 + 5 + n) * (H2 * W2D)]  = cls[n];
}

extern "C" void kernel_launch(void* const* d_in, const int* in_sizes, int n_in,
                              void* d_out, int out_size, void* d_ws, size_t ws_size,
                              hipStream_t stream) {
  const float* feat0 = (const float*)d_in[0];
  const float* feat1 = (const float*)d_in[1];
  const float* w1    = (const float*)d_in[2];
  const float* g1    = (const float*)d_in[3];
  const float* b1    = (const float*)d_in[4];
  const float* m1    = (const float*)d_in[5];
  const float* v1    = (const float*)d_in[6];
  const float* w2    = (const float*)d_in[7];
  const float* g2    = (const float*)d_in[8];
  const float* b2    = (const float*)d_in[9];
  const float* m2    = (const float*)d_in[10];
  const float* v2    = (const float*)d_in[11];
  const float* wd    = (const float*)d_in[12];
  const float* bd    = (const float*)d_in[13];

  char* ws = (char*)d_ws;
  _Float16* ft   = (_Float16*)(ws + OFF_FT);
  _Float16* cat  = (_Float16*)(ws + OFF_CAT);
  _Float16* w1p  = (_Float16*)(ws + OFF_W1P);
  _Float16* w2p  = (_Float16*)(ws + OFF_W2P);
  float*    pre  = (float*)   (ws + OFF_PRE);
  float*    out  = (float*)d_out;

  int n_t1 = BB * C1IN * H1 * W1;          // 11,829,248
  int n_t0 = BB * 1024 * (H2 * W2D);       //  5,914,624
  k_t1    <<<(n_t1 + 255) / 256, 256, 0, stream>>>(feat1, ft);
  k_t0    <<<(n_t0 + 255) / 256, 256, 0, stream>>>(feat0, cat);
  k_zero  <<<(C2IN + 255) / 256, 256, 0, stream>>>(cat);
  k_packw1<<<(4 * 16 * 32 + 255) / 256, 256, 0, stream>>>(w1, w1p);
  k_packw2<<<(64 * 9 * 40 * 32 + 255) / 256, 256, 0, stream>>>(w2, w2p);

  k_conv1 <<<NPIX1 / 16, 128, 0, stream>>>(ft, w1p, g1, b1, m1, v1, cat);
  k_conv2 <<<dim3(NPIX2 / 16, 2), 256, 0, stream>>>(cat, w2p, g2, b2, m2, v2, pre);
  k_decode<<<(NPIX2 * 5 + 127) / 128, 128, 0, stream>>>(pre, wd, bd, out);
}
